// MultiHeadAttention_24696061952589
// MI455X (gfx1250) — matrix operand order repaired and checked by reading, not compiled, not run on hardware
//
#include <hip/hip_runtime.h>
#include <cstdint>
#include <cstddef>

// Problem constants (match reference)
#define BB   2
#define SS   2048
#define DD   1024
#define HH   16
#define DH   64
#define NHID (HH * DH)          // 1024
#define MROWS (BB * SS)         // 4096 rows of the token-major matrices
#define INV_TEMP 0.125f         // 1/sqrt(64)

typedef __attribute__((ext_vector_type(16))) _Float16 v16h;
typedef __attribute__((ext_vector_type(8)))  _Float16 v8h;
typedef __attribute__((ext_vector_type(8)))  float    v8f;

static __device__ __forceinline__ v8f wmma_f16(v16h a, v16h b, v8f c) {
  // v_wmma_f32_16x16x32_f16  D = A*B + C   (8 args: neg_a,A,neg_b,B,c_mod,C,reuse_a,reuse_b)
  return __builtin_amdgcn_wmma_f32_16x16x32_f16(false, a, false, b, (short)0, c, false, false);
}

// ---- Fragment loaders -------------------------------------------------------
// A (16-bit, 16x32 MxK): lane (0..15) -> M=lane, K = {k0+kh+0..7, k0+kh+16..23}
//                        lane (16..31)-> M=lane-16, kh = 8
static __device__ __forceinline__ v16h load_fragA_h(const _Float16* __restrict__ p,
                                                    int ld, int m0, int k0, int lane) {
  int r  = m0 + (lane & 15);
  int kh = (lane >> 4) << 3;
  const _Float16* base = p + (size_t)r * ld + k0 + kh;
  v8h lo = *(const v8h*)(base);
  v8h hi = *(const v8h*)(base + 16);
  v16h f;
#pragma unroll
  for (int i = 0; i < 8; ++i) { f[i] = lo[i]; f[i + 8] = hi[i]; }
  return f;
}

// A fragment sourced from fp32 data (attention probabilities), converted to f16.
static __device__ __forceinline__ v16h load_fragA_f32(const float* __restrict__ p,
                                                      int ld, int m0, int k0, int lane) {
  int r  = m0 + (lane & 15);
  int kh = (lane >> 4) << 3;
  const float* base = p + (size_t)r * ld + k0 + kh;
  v16h f;
#pragma unroll
  for (int i = 0; i < 8; ++i) {
    f[i]     = (_Float16)base[i];
    f[i + 8] = (_Float16)base[16 + i];
  }
  return f;
}

// B (16-bit, 32x16 KxN) from [N,K]-contiguous storage: lane -> N = n0 + (lane&15),
// K = {k0+kh+0..7, k0+kh+16..23}, kh = 8*(lane>>4)
static __device__ __forceinline__ v16h load_fragB_h(const _Float16* __restrict__ p,
                                                    int ld, int n0, int k0, int lane) {
  int n  = n0 + (lane & 15);
  int kh = (lane >> 4) << 3;
  const _Float16* base = p + (size_t)n * ld + k0 + kh;
  v8h lo = *(const v8h*)(base);
  v8h hi = *(const v8h*)(base + 16);
  v16h f;
#pragma unroll
  for (int i = 0; i < 8; ++i) { f[i] = lo[i]; f[i + 8] = hi[i]; }
  return f;
}

// ---- Elementwise conversion kernels ----------------------------------------
__global__ void k_cvt_f16(const float* __restrict__ in, _Float16* __restrict__ out, int n) {
  int i = blockIdx.x * blockDim.x + threadIdx.x;
  if (i < n) out[i] = (_Float16)in[i];
}

// W [K=1024, N=1024] fp32 row-major  ->  WT [N, K] f16  (WT[n*1024+k] = W[k*1024+n])
__global__ void k_cvtT_f16(const float* __restrict__ in, _Float16* __restrict__ out) {
  int i = blockIdx.x * blockDim.x + threadIdx.x;   // 1M elements
  int n = i >> 10, k = i & 1023;
  out[i] = (_Float16)in[(size_t)k * 1024 + n];
}

// ---- QKV projection: X[4096,1024]f16 @ WT -> heads --------------------------
// mode 0/1: out[((b*H+h)*S+s)*64+dk]   (Q scaled by 1/TEMP via `scale`, K unscaled)
// mode 2  : out[((b*H+h)*64+dk)*S+s]   (V stored transposed for attn@V B-operand)
__global__ void k_proj(const _Float16* __restrict__ X, const _Float16* __restrict__ WT,
                       _Float16* __restrict__ out, float scale, int mode) {
  int wid  = (int)((blockIdx.x * (size_t)blockDim.x + threadIdx.x) >> 5);
  int lane = threadIdx.x & 31;
  int tn = wid & 31;            // 32 tiles across N=1024
  int tm = wid >> 5;            // 128 tiles across M=4096
  int m0 = tm << 5, n0 = tn << 5;

  v8f acc[2][2] = {};
  for (int k0 = 0; k0 < DD; k0 += 32) {
    v16h a0 = load_fragA_h(X, DD, m0,      k0, lane);
    v16h a1 = load_fragA_h(X, DD, m0 + 16, k0, lane);
    v16h b0 = load_fragB_h(WT, DD, n0,      k0, lane);
    v16h b1 = load_fragB_h(WT, DD, n0 + 16, k0, lane);
    acc[0][0] = wmma_f16(a0, b0, acc[0][0]);
    acc[0][1] = wmma_f16(a0, b1, acc[0][1]);
    acc[1][0] = wmma_f16(a1, b0, acc[1][0]);
    acc[1][1] = wmma_f16(a1, b1, acc[1][1]);
  }

  int rl = (lane >> 4) << 3, cl = lane & 15;
#pragma unroll
  for (int mi = 0; mi < 2; ++mi)
#pragma unroll
    for (int ni = 0; ni < 2; ++ni)
#pragma unroll
      for (int j = 0; j < 8; ++j) {
        int m = m0 + mi * 16 + j + rl;
        int n = n0 + ni * 16 + cl;
        int b = m >> 11, s = m & (SS - 1);
        int h = n >> 6,  dk = n & 63;
        float v = acc[mi][ni][j] * scale;
        size_t idx = (mode == 2)
                   ? ((size_t)((b * HH + h) * DH + dk)) * SS + s
                   : ((size_t)((b * HH + h) * SS + s)) * DH + dk;
        out[idx] = (_Float16)v;
      }
}

// ---- scores = Qh @ Kh^T  (+mask), raw fp32 into attn region -----------------
__global__ void k_scores(const _Float16* __restrict__ Qh, const _Float16* __restrict__ Kh,
                         const int* __restrict__ mask, float* __restrict__ attn) {
  int wid  = (int)((blockIdx.x * (size_t)blockDim.x + threadIdx.x) >> 5);
  int lane = threadIdx.x & 31;
  int bh = wid >> 12;               // 64*64 = 4096 tiles per (b,h)
  int t  = wid & 4095;
  int tm = t >> 6, tn = t & 63;
  int m0 = tm << 5, n0 = tn << 5;

  const _Float16* A  = Qh + (size_t)bh * SS * DH;
  const _Float16* Bp = Kh + (size_t)bh * SS * DH;   // [S,64] == [N,K] form of K^T
  int b = bh >> 4;

  v8f acc[2][2] = {};
#pragma unroll
  for (int k0 = 0; k0 < DH; k0 += 32) {
    v16h a0 = load_fragA_h(A, DH, m0,      k0, lane);
    v16h a1 = load_fragA_h(A, DH, m0 + 16, k0, lane);
    v16h b0 = load_fragB_h(Bp, DH, n0,      k0, lane);
    v16h b1 = load_fragB_h(Bp, DH, n0 + 16, k0, lane);
    acc[0][0] = wmma_f16(a0, b0, acc[0][0]);
    acc[0][1] = wmma_f16(a0, b1, acc[0][1]);
    acc[1][0] = wmma_f16(a1, b0, acc[1][0]);
    acc[1][1] = wmma_f16(a1, b1, acc[1][1]);
  }

  const int* mrow = mask + (size_t)b * SS * SS;
  float* out = attn + (size_t)bh * SS * SS;
  int rl = (lane >> 4) << 3, cl = lane & 15;
#pragma unroll
  for (int mi = 0; mi < 2; ++mi)
#pragma unroll
    for (int ni = 0; ni < 2; ++ni)
#pragma unroll
      for (int j = 0; j < 8; ++j) {
        int sq = m0 + mi * 16 + j + rl;
        int sk = n0 + ni * 16 + cl;
        float v = acc[mi][ni][j];
        if (mrow[(size_t)sq * SS + sk] == 0) v = -1e9f;
        out[(size_t)sq * SS + sk] = v;
      }
}

// ---- in-place row softmax over attn (row length 2048) -----------------------
__global__ void k_softmax(float* __restrict__ attn) {
  __shared__ float red[256];
  int t = threadIdx.x;
  float* p = attn + (size_t)blockIdx.x * SS;
  float4 v0 = *(float4*)(p + t * 8);
  float4 v1 = *(float4*)(p + t * 8 + 4);
  float m = fmaxf(fmaxf(fmaxf(v0.x, v0.y), fmaxf(v0.z, v0.w)),
                  fmaxf(fmaxf(v1.x, v1.y), fmaxf(v1.z, v1.w)));
  red[t] = m; __syncthreads();
  for (int off = 128; off > 0; off >>= 1) {
    if (t < off) red[t] = fmaxf(red[t], red[t + off]);
    __syncthreads();
  }
  float rmax = red[0]; __syncthreads();
  v0.x = __expf(v0.x - rmax); v0.y = __expf(v0.y - rmax);
  v0.z = __expf(v0.z - rmax); v0.w = __expf(v0.w - rmax);
  v1.x = __expf(v1.x - rmax); v1.y = __expf(v1.y - rmax);
  v1.z = __expf(v1.z - rmax); v1.w = __expf(v1.w - rmax);
  float s = v0.x + v0.y + v0.z + v0.w + v1.x + v1.y + v1.z + v1.w;
  red[t] = s; __syncthreads();
  for (int off = 128; off > 0; off >>= 1) {
    if (t < off) red[t] += red[t + off];
    __syncthreads();
  }
  float inv = 1.0f / red[0];
  v0.x *= inv; v0.y *= inv; v0.z *= inv; v0.w *= inv;
  v1.x *= inv; v1.y *= inv; v1.z *= inv; v1.w *= inv;
  *(float4*)(p + t * 8)     = v0;
  *(float4*)(p + t * 8 + 4) = v1;
}

// ---- O = attn @ V   (A fp32->f16 on load, Vt is [B,H,64,S] f16) -------------
__global__ void k_attnv(const float* __restrict__ attn, const _Float16* __restrict__ Vt,
                        _Float16* __restrict__ O) {
  int wid  = (int)((blockIdx.x * (size_t)blockDim.x + threadIdx.x) >> 5);
  int lane = threadIdx.x & 31;
  int bh = wid >> 7;                // 64*2 = 128 tiles per (b,h)
  int t  = wid & 127;
  int tm = t >> 1, tn = t & 1;
  int m0 = tm << 5, n0 = tn << 5;

  const float*    A  = attn + (size_t)bh * SS * SS;
  const _Float16* Bp = Vt   + (size_t)bh * DH * SS;   // [64,S] == [N,K]

  v8f acc[2][2] = {};
  for (int k0 = 0; k0 < SS; k0 += 32) {
    v16h a0 = load_fragA_f32(A, SS, m0,      k0, lane);
    v16h a1 = load_fragA_f32(A, SS, m0 + 16, k0, lane);
    v16h b0 = load_fragB_h(Bp, SS, n0,      k0, lane);
    v16h b1 = load_fragB_h(Bp, SS, n0 + 16, k0, lane);
    acc[0][0] = wmma_f16(a0, b0, acc[0][0]);
    acc[0][1] = wmma_f16(a0, b1, acc[0][1]);
    acc[1][0] = wmma_f16(a1, b0, acc[1][0]);
    acc[1][1] = wmma_f16(a1, b1, acc[1][1]);
  }

  int b = bh >> 4, h = bh & 15;
  int rl = (lane >> 4) << 3, cl = lane & 15;
#pragma unroll
  for (int mi = 0; mi < 2; ++mi)
#pragma unroll
    for (int ni = 0; ni < 2; ++ni)
#pragma unroll
      for (int j = 0; j < 8; ++j) {
        int s  = m0 + mi * 16 + j + rl;
        int dv = n0 + ni * 16 + cl;
        O[((size_t)(b * SS + s)) * NHID + h * DH + dv] = (_Float16)acc[mi][ni][j];
      }
}

// ---- X = O @ Wo + residual (fp32 out) ---------------------------------------
__global__ void k_outproj(const _Float16* __restrict__ O, const _Float16* __restrict__ WoT,
                          const float* __restrict__ resid, float* __restrict__ X) {
  int wid  = (int)((blockIdx.x * (size_t)blockDim.x + threadIdx.x) >> 5);
  int lane = threadIdx.x & 31;
  int tn = wid & 31;
  int tm = wid >> 5;
  int m0 = tm << 5, n0 = tn << 5;

  v8f acc[2][2] = {};
  for (int k0 = 0; k0 < NHID; k0 += 32) {
    v16h a0 = load_fragA_h(O, NHID, m0,      k0, lane);
    v16h a1 = load_fragA_h(O, NHID, m0 + 16, k0, lane);
    v16h b0 = load_fragB_h(WoT, NHID, n0,      k0, lane);
    v16h b1 = load_fragB_h(WoT, NHID, n0 + 16, k0, lane);
    acc[0][0] = wmma_f16(a0, b0, acc[0][0]);
    acc[0][1] = wmma_f16(a0, b1, acc[0][1]);
    acc[1][0] = wmma_f16(a1, b0, acc[1][0]);
    acc[1][1] = wmma_f16(a1, b1, acc[1][1]);
  }

  int rl = (lane >> 4) << 3, cl = lane & 15;
#pragma unroll
  for (int mi = 0; mi < 2; ++mi)
#pragma unroll
    for (int ni = 0; ni < 2; ++ni)
#pragma unroll
      for (int j = 0; j < 8; ++j) {
        int m = m0 + mi * 16 + j + rl;
        int n = n0 + ni * 16 + cl;
        size_t idx = (size_t)m * DD + n;
        X[idx] = acc[mi][ni][j] + resid[idx];
      }
}

// ---- LayerNorm over D=1024 --------------------------------------------------
__global__ void k_ln(const float* __restrict__ X, const float* __restrict__ gamma,
                     const float* __restrict__ beta, float* __restrict__ out) {
  __shared__ float rs[256];
  __shared__ float rs2[256];
  int t = threadIdx.x;
  const float* p = X + (size_t)blockIdx.x * DD;
  float4 v = *(const float4*)(p + t * 4);
  rs[t]  = v.x + v.y + v.z + v.w;
  rs2[t] = v.x * v.x + v.y * v.y + v.z * v.z + v.w * v.w;
  __syncthreads();
  for (int off = 128; off > 0; off >>= 1) {
    if (t < off) { rs[t] += rs[t + off]; rs2[t] += rs2[t + off]; }
    __syncthreads();
  }
  float mu  = rs[0] * (1.0f / DD);
  float var = rs2[0] * (1.0f / DD) - mu * mu;
  float inv = rsqrtf(var + 1e-6f);
  float4 g = *(const float4*)(gamma + t * 4);
  float4 bta = *(const float4*)(beta + t * 4);
  float4 o;
  o.x = (v.x - mu) * inv * g.x + bta.x;
  o.y = (v.y - mu) * inv * g.y + bta.y;
  o.z = (v.z - mu) * inv * g.z + bta.z;
  o.w = (v.w - mu) * inv * g.w + bta.w;
  *(float4*)(out + (size_t)blockIdx.x * DD + t * 4) = o;
}

// -----------------------------------------------------------------------------
extern "C" void kernel_launch(void* const* d_in, const int* in_sizes, int n_in,
                              void* d_out, int out_size, void* d_ws, size_t ws_size,
                              hipStream_t stream) {
  (void)in_sizes; (void)n_in; (void)out_size; (void)ws_size;
  const float* q    = (const float*)d_in[0];
  const float* k    = (const float*)d_in[1];
  const float* v    = (const float*)d_in[2];
  const int*   mask = (const int*)d_in[3];
  const float* Wq   = (const float*)d_in[4];
  const float* Wk   = (const float*)d_in[5];
  const float* Wv   = (const float*)d_in[6];
  const float* Wo   = (const float*)d_in[7];
  const float* lng  = (const float*)d_in[8];
  const float* lnb  = (const float*)d_in[9];

  const size_t nTok = (size_t)MROWS * DD;       // 4,194,304
  const size_t nW   = (size_t)DD * NHID;        // 1,048,576

  // Workspace layout (bytes)
  char* ws = (char*)d_ws;
  size_t off = 0;
  _Float16* qf = (_Float16*)(ws + off); off += nTok * 2;
  _Float16* kf = (_Float16*)(ws + off); off += nTok * 2;
  _Float16* vf = (_Float16*)(ws + off); off += nTok * 2;
  _Float16* wqT = (_Float16*)(ws + off); off += nW * 2;
  _Float16* wkT = (_Float16*)(ws + off); off += nW * 2;
  _Float16* wvT = (_Float16*)(ws + off); off += nW * 2;
  _Float16* woT = (_Float16*)(ws + off); off += nW * 2;
  _Float16* Qh = (_Float16*)(ws + off); off += nTok * 2;
  _Float16* Kh = (_Float16*)(ws + off); off += nTok * 2;
  _Float16* Vt = (_Float16*)(ws + off); off += nTok * 2;
  _Float16* O  = (_Float16*)(ws + off); off += nTok * 2;
  float*    xb = (float*)(ws + off);    off += nTok * 4;

  float* x_out    = (float*)d_out;                 // [B,S,D]
  float* attn_out = x_out + nTok;                  // [B,H,S,S]

  // 1) fp32 -> f16 activations
  k_cvt_f16<<<16384, 256, 0, stream>>>(q, qf, (int)nTok);
  k_cvt_f16<<<16384, 256, 0, stream>>>(k, kf, (int)nTok);
  k_cvt_f16<<<16384, 256, 0, stream>>>(v, vf, (int)nTok);
  // 2) weights -> f16 transposed [N,K]
  k_cvtT_f16<<<4096, 256, 0, stream>>>(Wq, wqT);
  k_cvtT_f16<<<4096, 256, 0, stream>>>(Wk, wkT);
  k_cvtT_f16<<<4096, 256, 0, stream>>>(Wv, wvT);
  k_cvtT_f16<<<4096, 256, 0, stream>>>(Wo, woT);
  // 3) QKV projections (4096 waves each -> 512 blocks of 8 waves)
  k_proj<<<512, 256, 0, stream>>>(qf, wqT, Qh, INV_TEMP, 0);
  k_proj<<<512, 256, 0, stream>>>(kf, wkT, Kh, 1.0f, 1);
  k_proj<<<512, 256, 0, stream>>>(vf, wvT, Vt, 1.0f, 2);
  // 4) scores + mask (B*H*64*64 = 131072 waves -> 16384 blocks)
  k_scores<<<16384, 256, 0, stream>>>(Qh, Kh, mask, attn_out);
  // 5) row softmax in place (B*H*S rows)
  k_softmax<<<BB * HH * SS, 256, 0, stream>>>(attn_out);
  // 6) attn @ V (4096 waves -> 512 blocks)
  k_attnv<<<512, 256, 0, stream>>>(attn_out, Vt, O);
  // 7) output projection + residual
  k_outproj<<<512, 256, 0, stream>>>(O, woT, q, xb);
  // 8) LayerNorm -> x output
  k_ln<<<MROWS, 256, 0, stream>>>(xb, lng, lnb, x_out);
}
